// GroupedQueryAttention_56427280335137
// MI455X (gfx1250) — compile-verified, matrix operand order checked by reading, not run on hardware
//
#include <hip/hip_runtime.h>
#include <cstdint>

// ---------------------------------------------------------------------------
// GQA forward for MI455X (gfx1250, wave32, WMMA + async-to-LDS).
// Pipeline: cvt(f32->bf16) -> WMMA GEMM (Q,K,V proj) -> RoPE+cvt ->
//           flash-attention (WMMA QK^T / PV, online softmax) -> WMMA GEMM (Wo)
// Data movement: GLOBAL_LOAD_ASYNC_TO_LDS_B128 for straight-copy tiles,
//                VGPR round-trip only where a transpose is required.
// ---------------------------------------------------------------------------

#define B_SZ 2
#define T_SZ 2048
#define DIN  2048
#define DOUT 2048
#define NH   32
#define NKV  8
#define HD   64
#define ROWS (B_SZ * T_SZ)   // 4096
#define KVW  (NKV * HD)      // 512

typedef unsigned short u16;
typedef __bf16 bf16;
typedef bf16  v16bf __attribute__((ext_vector_type(16)));
typedef float v8f   __attribute__((ext_vector_type(8)));
typedef u16   v16u  __attribute__((ext_vector_type(16)));
typedef u16   v8u   __attribute__((ext_vector_type(8)));

__device__ __forceinline__ u16 f2bf(float f) {
  unsigned u = __builtin_bit_cast(unsigned, f);
  u += 0x7FFFu + ((u >> 16) & 1u);            // round-to-nearest-even
  return (u16)(u >> 16);
}

// Async copy of 16 bytes/lane from global memory straight into LDS.
// ldsByteOff is the byte offset within the workgroup's LDS allocation
// (single __shared__ arena per kernel -> arena base is offset 0).
__device__ __forceinline__ void async_cp16(unsigned ldsByteOff, const void* gp) {
  asm volatile("global_load_async_to_lds_b128 %0, %1, off"
               :: "v"(ldsByteOff), "v"((unsigned long long)(uintptr_t)gp)
               : "memory");
}
__device__ __forceinline__ void wait_async() {
  asm volatile("s_wait_asynccnt 0x0" ::: "memory");
}

// Load a 16x32 bf16 A/B fragment row from LDS (row-major, padded stride).
// ISA layout (16-bit A 16x32): lanes 0-15 -> elems K{0..7,16..23},
// lanes 16-31 -> K{8..15,24..31}; B operand mirrors with lane = N column.
__device__ __forceinline__ v16bf frag_ld(const u16* rowp, int k0, int half) {
  const u16* q = rowp + k0 + half * 8;
  v8u lo = *(const v8u*)(q);
  v8u hi = *(const v8u*)(q + 16);
  v16u r;
#pragma unroll
  for (int e = 0; e < 8; ++e) { r[e] = lo[e]; r[e + 8] = hi[e]; }
  union { v16u u; v16bf b; } cv; cv.u = r;
  return cv.b;
}

// ------------------------------- cvt ---------------------------------------
__global__ void cvt_f32_bf16(const float* __restrict__ in, u16* __restrict__ out, int n) {
  int i = blockIdx.x * blockDim.x + threadIdx.x;
  if (i < n) out[i] = f2bf(in[i]);
}

// ------------------------------- RoPE --------------------------------------
__global__ void rope_kernel(const float* __restrict__ in, u16* __restrict__ out,
                            int heads, int width) {
  int idx = blockIdx.x * blockDim.x + threadIdx.x;
  int ppr = heads * 32;
  int row = idx / ppr;
  int rem = idx - row * ppr;
  int hh = rem >> 5;
  int d  = rem & 31;
  int t  = row & (T_SZ - 1);
  float theta = __powf(10000.0f, -(float)d * (1.0f / 32.0f));
  float fr = (float)t * theta;
  float s, c;
  __sincosf(fr, &s, &c);
  size_t base = (size_t)row * width + hh * HD + d;
  float x1 = in[base], x2 = in[base + 32];
  out[base]      = f2bf(x1 * c - x2 * s);
  out[base + 32] = f2bf(x2 * c + x1 * s);
}

// ------------------------------- GEMM --------------------------------------
// C[M,N] = A[M,K] @ B[K,N]; A,B bf16 row-major, C f32. BM=64 BN=128 BK=32.
// 8 waves: 2x4 wave grid, each wave a 32x32 block (2x2 WMMA tiles).
// A tile: async copy global->LDS. B tile: VGPR round-trip (transposed store).
#define BM 64
#define BN 128
#define BK 32
#define ASTR 40
#define BSTR 40
#define G_AS 0                 // As[r][k] at element (r*ASTR + k)
#define G_BS (BM * ASTR)       // Bs[n][k] at element (G_BS + n*BSTR + k)

__global__ __launch_bounds__(256) void gemm_bf16(
    const u16* __restrict__ A, const u16* __restrict__ Bm, float* __restrict__ C,
    int M, int N, int K) {
  __shared__ u16 sm[G_BS + BN * BSTR];
  const int t = threadIdx.x;
  const int wid = t >> 5, lane = t & 31, l15 = lane & 15, half = lane >> 4;
  const int wm = (wid & 1) * 32, wn = (wid >> 1) * 32;
  const int rowBase = blockIdx.y * BM;
  const int colBase = blockIdx.x * BN;
  v8f acc[2][2] = {};
  const int arow = t >> 2, ak = (t & 3) * 8;

  for (int k0 = 0; k0 < K; k0 += BK) {
    // A tile: straight copy -> async to LDS (b128, one lane = one 16B row chunk)
    async_cp16((unsigned)((G_AS + arow * ASTR + ak) * 2),
               &A[(size_t)(rowBase + arow) * K + k0 + ak]);
    // B tile: transpose via VGPRs
#pragma unroll
    for (int it = 0; it < 2; ++it) {
      int lin = t + it * 256;
      int kk = lin >> 4, n8 = (lin & 15) * 8;
      v8u v = *(const v8u*)&Bm[(size_t)(k0 + kk) * N + colBase + n8];
#pragma unroll
      for (int j = 0; j < 8; ++j) sm[G_BS + (n8 + j) * BSTR + kk] = v[j];
    }
    if (k0 + BK < K) {  // warm L2 for the next k-step
      __builtin_prefetch(&A[(size_t)(rowBase + arow) * K + k0 + BK + ak], 0, 0);
      __builtin_prefetch(&Bm[(size_t)(k0 + BK + (t >> 4)) * N + colBase + (t & 15) * 8], 0, 0);
    }
    wait_async();
    __syncthreads();

    v16bf afr[2], bfr[2];
#pragma unroll
    for (int mt = 0; mt < 2; ++mt)
      afr[mt] = frag_ld(&sm[G_AS + (wm + mt * 16 + l15) * ASTR], 0, half);
#pragma unroll
    for (int nt = 0; nt < 2; ++nt)
      bfr[nt] = frag_ld(&sm[G_BS + (wn + nt * 16 + l15) * BSTR], 0, half);
#pragma unroll
    for (int mt = 0; mt < 2; ++mt)
#pragma unroll
      for (int nt = 0; nt < 2; ++nt)
        acc[mt][nt] = __builtin_amdgcn_wmma_f32_16x16x32_bf16(
            false, afr[mt], false, bfr[nt], (short)0, acc[mt][nt], false, false);
    __syncthreads();
  }

#pragma unroll
  for (int mt = 0; mt < 2; ++mt)
#pragma unroll
    for (int nt = 0; nt < 2; ++nt)
#pragma unroll
      for (int i = 0; i < 8; ++i) {
        int r = rowBase + wm + mt * 16 + (half ? i + 8 : i);
        int c = colBase + wn + nt * 16 + l15;
        C[(size_t)r * N + c] = acc[mt][nt][i];
      }
}

// --------------------------- flash attention -------------------------------
// Block = (b, h, 64-query tile), 4 waves, each wave owns 16 query rows.
// Q/K tiles: async copy global->LDS. V tile: transposed via VGPRs.
#define QSTR 72
#define F_Q  0
#define F_K  (64 * QSTR)
#define F_V  (128 * QSTR)
#define F_P  (192 * QSTR)      // + w*16*QSTR per wave

__global__ __launch_bounds__(128) void fattn(
    const u16* __restrict__ Qb, const u16* __restrict__ Kb,
    const u16* __restrict__ Vb, u16* __restrict__ Ctx) {
  __shared__ u16 sm[F_P + 4 * 16 * QSTR];
  const int t = threadIdx.x;
  const int w = t >> 5, lane = t & 31, l15 = lane & 15, half = lane >> 4;
  const int bh = blockIdx.x;
  const int b = bh >> 5, h = bh & 31, kvh = h >> 2;
  const int qb = blockIdx.y;
  const size_t qrowBase = (size_t)b * T_SZ + (size_t)qb * 64;

#pragma unroll
  for (int it = 0; it < 4; ++it) {
    int lin = t + it * 128;
    int r = lin >> 3, d8 = (lin & 7) * 8;
    async_cp16((unsigned)((F_Q + r * QSTR + d8) * 2),
               &Qb[(qrowBase + r) * DOUT + h * HD + d8]);
  }

  float m[8], l[8];
#pragma unroll
  for (int i = 0; i < 8; ++i) { m[i] = -1e30f; l[i] = 0.f; }
  v8f acc[4] = {};
  const float scale = 0.125f;  // 1/sqrt(64)

  for (int kb = 0; kb <= qb; ++kb) {
    const size_t krowBase = (size_t)b * T_SZ + (size_t)kb * 64;
#pragma unroll
    for (int it = 0; it < 4; ++it) {
      int lin = t + it * 128;
      int r = lin >> 3, d8 = (lin & 7) * 8;
      async_cp16((unsigned)((F_K + r * QSTR + d8) * 2),
                 &Kb[(krowBase + r) * KVW + kvh * HD + d8]);
      v8u v = *(const v8u*)&Vb[(krowBase + r) * KVW + kvh * HD + d8];
#pragma unroll
      for (int j = 0; j < 8; ++j) sm[F_V + (d8 + j) * QSTR + r] = v[j];
    }
    wait_async();
    __syncthreads();

    // S = Q K^T (per-wave 16x64)
    v8f sf[4] = {};
    v16bf aq[2];
#pragma unroll
    for (int kk = 0; kk < 2; ++kk)
      aq[kk] = frag_ld(&sm[F_Q + (w * 16 + l15) * QSTR], kk * 32, half);
#pragma unroll
    for (int nt = 0; nt < 4; ++nt)
#pragma unroll
      for (int kk = 0; kk < 2; ++kk) {
        v16bf bk = frag_ld(&sm[F_K + (nt * 16 + l15) * QSTR], kk * 32, half);
        sf[nt] = __builtin_amdgcn_wmma_f32_16x16x32_bf16(
            false, aq[kk], false, bk, (short)0, sf[nt], false, false);
      }

    // scale + causal mask + row max
    const bool diag = (kb == qb);
    float mloc[8];
#pragma unroll
    for (int i = 0; i < 8; ++i) mloc[i] = -1e30f;
#pragma unroll
    for (int nt = 0; nt < 4; ++nt) {
      int col = nt * 16 + l15;
#pragma unroll
      for (int i = 0; i < 8; ++i) {
        int r = half ? i + 8 : i;
        float sv = sf[nt][i] * scale;
        if (diag && col > w * 16 + r) sv = -1e30f;
        sf[nt][i] = sv;
        mloc[i] = fmaxf(mloc[i], sv);
      }
    }
#pragma unroll
    for (int off = 1; off < 16; off <<= 1)
#pragma unroll
      for (int i = 0; i < 8; ++i)
        mloc[i] = fmaxf(mloc[i], __shfl_xor(mloc[i], off, 32));

    float alpha[8], rs[8];
#pragma unroll
    for (int i = 0; i < 8; ++i) {
      float mn = fmaxf(m[i], mloc[i]);
      alpha[i] = __expf(m[i] - mn);
      m[i] = mn;
      rs[i] = 0.f;
    }
#pragma unroll
    for (int nt = 0; nt < 4; ++nt)
#pragma unroll
      for (int i = 0; i < 8; ++i) {
        float p = __expf(sf[nt][i] - m[i]);
        sf[nt][i] = p;
        rs[i] += p;
      }
#pragma unroll
    for (int off = 1; off < 16; off <<= 1)
#pragma unroll
      for (int i = 0; i < 8; ++i) rs[i] += __shfl_xor(rs[i], off, 32);
#pragma unroll
    for (int i = 0; i < 8; ++i) l[i] = l[i] * alpha[i] + rs[i];
#pragma unroll
    for (int nt = 0; nt < 4; ++nt)
#pragma unroll
      for (int i = 0; i < 8; ++i) acc[nt][i] *= alpha[i];

    // stage P (bf16) in per-wave LDS to re-fragment as WMMA A operand
#pragma unroll
    for (int nt = 0; nt < 4; ++nt) {
      int col = nt * 16 + l15;
#pragma unroll
      for (int i = 0; i < 8; ++i) {
        int r = half ? i + 8 : i;
        sm[F_P + (w * 16 + r) * QSTR + col] = f2bf(sf[nt][i]);
      }
    }

    // O += P @ V  (Vt[d][s] gives the B operand column-major over keys)
#pragma unroll
    for (int kk = 0; kk < 2; ++kk) {
      v16bf pa = frag_ld(&sm[F_P + (w * 16 + l15) * QSTR], kk * 32, half);
#pragma unroll
      for (int nt = 0; nt < 4; ++nt) {
        v16bf vb = frag_ld(&sm[F_V + (nt * 16 + l15) * QSTR], kk * 32, half);
        acc[nt] = __builtin_amdgcn_wmma_f32_16x16x32_bf16(
            false, pa, false, vb, (short)0, acc[nt], false, false);
      }
    }
    __syncthreads();
  }

  // normalize and write ctx[b][t][h*64+d] as bf16
#pragma unroll
  for (int nt = 0; nt < 4; ++nt) {
    int col = h * HD + nt * 16 + l15;
#pragma unroll
    for (int i = 0; i < 8; ++i) {
      int r = half ? i + 8 : i;
      Ctx[(qrowBase + w * 16 + r) * DOUT + col] = f2bf(acc[nt][i] / l[i]);
    }
  }
}

// ------------------------------ launch -------------------------------------
extern "C" void kernel_launch(void* const* d_in, const int* in_sizes, int n_in,
                              void* d_out, int out_size, void* d_ws, size_t ws_size,
                              hipStream_t stream) {
  (void)in_sizes; (void)n_in; (void)out_size; (void)ws_size;
  const float* x  = (const float*)d_in[0];
  const float* Wq = (const float*)d_in[1];
  const float* Wk = (const float*)d_in[2];
  const float* Wv = (const float*)d_in[3];
  const float* Wo = (const float*)d_in[4];
  float* out = (float*)d_out;

  char* ws = (char*)d_ws;
  size_t off = 0;
  auto alloc = [&](size_t bytes) -> void* {
    void* p = ws + off;
    off += (bytes + 255) & ~(size_t)255;
    return p;
  };
  u16*   xb  = (u16*)alloc((size_t)ROWS * DIN * 2);
  u16*   Wqb = (u16*)alloc((size_t)DIN * DOUT * 2);
  u16*   Wkb = (u16*)alloc((size_t)DIN * KVW * 2);
  u16*   Wvb = (u16*)alloc((size_t)DIN * KVW * 2);
  u16*   Wob = (u16*)alloc((size_t)DOUT * DOUT * 2);
  float* Qf  = (float*)alloc((size_t)ROWS * DOUT * 4);
  float* Kf  = (float*)alloc((size_t)ROWS * KVW * 4);
  float* Vf  = (float*)alloc((size_t)ROWS * KVW * 4);
  u16*   Qr  = (u16*)alloc((size_t)ROWS * DOUT * 2);
  u16*   Kr  = (u16*)alloc((size_t)ROWS * KVW * 2);
  u16*   Vr  = (u16*)alloc((size_t)ROWS * KVW * 2);
  u16*   Ctx = (u16*)alloc((size_t)ROWS * DOUT * 2);

  // f32 -> bf16
  cvt_f32_bf16<<<(ROWS * DIN) / 256, 256, 0, stream>>>(x, xb, ROWS * DIN);
  cvt_f32_bf16<<<(DIN * DOUT) / 256, 256, 0, stream>>>(Wq, Wqb, DIN * DOUT);
  cvt_f32_bf16<<<(DIN * KVW) / 256, 256, 0, stream>>>(Wk, Wkb, DIN * KVW);
  cvt_f32_bf16<<<(DIN * KVW) / 256, 256, 0, stream>>>(Wv, Wvb, DIN * KVW);
  cvt_f32_bf16<<<(DOUT * DOUT) / 256, 256, 0, stream>>>(Wo, Wob, DOUT * DOUT);

  // projections
  gemm_bf16<<<dim3(DOUT / BN, ROWS / BM), 256, 0, stream>>>(xb, Wqb, Qf, ROWS, DOUT, DIN);
  gemm_bf16<<<dim3(KVW / BN, ROWS / BM), 256, 0, stream>>>(xb, Wkb, Kf, ROWS, KVW, DIN);
  gemm_bf16<<<dim3(KVW / BN, ROWS / BM), 256, 0, stream>>>(xb, Wvb, Vf, ROWS, KVW, DIN);

  // RoPE (Q,K) + bf16, plain bf16 for V
  rope_kernel<<<(ROWS * NH * 32) / 256, 256, 0, stream>>>(Qf, Qr, NH, DOUT);
  rope_kernel<<<(ROWS * NKV * 32) / 256, 256, 0, stream>>>(Kf, Kr, NKV, KVW);
  cvt_f32_bf16<<<(ROWS * KVW) / 256, 256, 0, stream>>>(Vf, Vr, ROWS * KVW);

  // attention
  fattn<<<dim3(B_SZ * NH, T_SZ / 64), 128, 0, stream>>>(Qr, Kr, Vr, Ctx);

  // output projection -> f32 d_out
  gemm_bf16<<<dim3(DOUT / BN, ROWS / BM), 256, 0, stream>>>(Ctx, Wob, out, ROWS, DOUT, DOUT);
}
